// AdvResNet_88433376625404
// MI455X (gfx1250) — matrix operand order repaired and checked by reading, not compiled, not run on hardware
//
#include <hip/hip_runtime.h>
#include <hip/hip_bf16.h>
#include <stdint.h>

// ---- problem constants (from reference) ----
#define EPSILON   0.1f
#define N_HID     4096
#define BATCH     4096
#define N_IN      2048
#define N_OUT     1024

// ---- tiling ----
#define BM 128
#define BN 128
#define BK 32
#define LDA 36   // f32 per A-tile row (32 + 4 pad): 144B rows -> 16B aligned, odd bank stride
#define LDB 36   // f32 per B-tile (transposed) row

#define AS1 __attribute__((address_space(1)))
#define AS3 __attribute__((address_space(3)))

typedef __attribute__((ext_vector_type(16))) __bf16 v16bf;
typedef __attribute__((ext_vector_type(8)))  float  v8f;
typedef int v4i_gcc __attribute__((vector_size(4 * sizeof(int))));  // matches builtin param

// ---------------- CDNA5 async global->LDS helpers (guarded) ------------------
__device__ __forceinline__ void async_b128(const void* g, void* l) {
#if __has_builtin(__builtin_amdgcn_global_load_async_to_lds_b128)
  __builtin_amdgcn_global_load_async_to_lds_b128(
      (AS1 v4i_gcc*)(uintptr_t)g, (AS3 v4i_gcc*)l, 0, 0);
#else
  uint32_t loff = (uint32_t)(uintptr_t)(AS3 void*)l;
  asm volatile("global_load_async_to_lds_b128 %0, %1, off"
               :: "v"(loff), "v"((uint64_t)(uintptr_t)g) : "memory");
#endif
}
__device__ __forceinline__ void async_b32(const void* g, void* l) {
#if __has_builtin(__builtin_amdgcn_global_load_async_to_lds_b32)
  __builtin_amdgcn_global_load_async_to_lds_b32(
      (AS1 int*)(uintptr_t)g, (AS3 int*)l, 0, 0);
#else
  uint32_t loff = (uint32_t)(uintptr_t)(AS3 void*)l;
  asm volatile("global_load_async_to_lds_b32 %0, %1, off"
               :: "v"(loff), "v"((uint64_t)(uintptr_t)g) : "memory");
#endif
}
__device__ __forceinline__ void wait_async_20() {  // one tile batch (20 ops) may remain
#if __has_builtin(__builtin_amdgcn_s_wait_asynccnt)
  __builtin_amdgcn_s_wait_asynccnt(20);
#else
  asm volatile("s_wait_asynccnt 20" ::: "memory");
#endif
}
__device__ __forceinline__ void wait_async_0() {
#if __has_builtin(__builtin_amdgcn_s_wait_asynccnt)
  __builtin_amdgcn_s_wait_asynccnt(0);
#else
  asm volatile("s_wait_asynccnt 0" ::: "memory");
#endif
}

// ---------------- precompute: beta_norm[n] = sum_k |beta[k,n]| ----------------
__global__ void beta_norm_kernel(const float* __restrict__ beta,
                                 float* __restrict__ ws) {
  int n = blockIdx.x * blockDim.x + threadIdx.x;
  if (n >= N_OUT) return;
  float s = 0.f;
  for (int k = 0; k < N_IN; ++k)             // coalesced across lanes
    s += fabsf(beta[(size_t)k * N_OUT + n]);
  ws[n] = s;
}

// ---------------- precompute: one[n] = N_HID * sum_j W2[n,j] + bias2[n] ------
__global__ void one_kernel(const float* __restrict__ W2,
                           const float* __restrict__ bias2,
                           float* __restrict__ ws) {
  __shared__ float red[256];
  const int n = blockIdx.x;
  float s = 0.f;
  for (int j = threadIdx.x; j < N_HID; j += 256)
    s += W2[(size_t)n * N_HID + j];
  red[threadIdx.x] = s;
  __syncthreads();
  for (int off = 128; off > 0; off >>= 1) {
    if (threadIdx.x < off) red[threadIdx.x] += red[threadIdx.x + off];
    __syncthreads();
  }
  if (threadIdx.x == 0) ws[N_OUT + n] = (float)N_HID * red[0] + bias2[n];
}

// ---------------- main fused GEMM + epilogue ---------------------------------
// out[b,n] = (x@beta)[b,n] + bias_lin[n] - EPS*y[b,n]*beta_norm[n] + one[n]
__global__ void __launch_bounds__(256)
adv_gemm_kernel(const float* __restrict__ x,
                const float* __restrict__ y,
                const float* __restrict__ beta,
                const float* __restrict__ bias_lin,
                const float* __restrict__ prec,   // [0..1023]=beta_norm, [1024..2047]=one
                float* __restrict__ out) {
  // f32 tiles in LDS, filled by async DMA; bf16 conversion happens at fragment build.
  __shared__ float Af[2][BM][LDA];   // A tile, row-major [m][k]
  __shared__ float Bf[2][BN][LDB];   // B tile, TRANSPOSED [n][k] (scatter during async copy)

  const int tid   = threadIdx.x;
  const int lane  = tid & 31;
  const int lo    = lane & 15;
  const int hi    = lane >> 4;
  const int wave  = tid >> 5;     // 0..7
  const int waveM = wave >> 1;    // 0..3  (32-row strip)
  const int waveN = wave & 1;     // 0..1  (64-col strip)

  const int rowBase = blockIdx.y * BM;
  const int colBase = blockIdx.x * BN;

  // issue one k-tile's async copies (20 ops per wave: 4xB128 for A, 16xB32 for B)
  auto issue = [&](int kt, int b) {
    const int k0 = kt * BK;
#pragma unroll
    for (int i = 0; i < 4; ++i) {                     // A: 128x32 f32, row-contiguous
      int idx = wave * 128 + i * 32 + lane;           // float4 index over [128][8]
      int r = idx >> 3, k4 = idx & 7;
      async_b128(x + (size_t)(rowBase + r) * N_IN + (k0 + k4 * 4),
                 &Af[b][r][k4 * 4]);
    }
#pragma unroll
    for (int i = 0; i < 16; ++i) {                    // B: 32x128 f32, transpose-scatter
      int idx = wave * 512 + i * 32 + lane;           // element index over [32][128]
      int bk = idx >> 7, bn = idx & 127;              // coalesced global (consecutive n)
      async_b32(beta + (size_t)(k0 + bk) * N_OUT + (colBase + bn),
                &Bf[b][bn][bk]);
    }
  };

  // ---- prologue: tile 0 -> buffer 0 ----
  issue(0, 0);

  v8f acc[2][4];
#pragma unroll
  for (int tm = 0; tm < 2; ++tm)
#pragma unroll
    for (int tn = 0; tn < 4; ++tn)
      acc[tm][tn] = (v8f){0.f, 0.f, 0.f, 0.f, 0.f, 0.f, 0.f, 0.f};

  const int KT = N_IN / BK;   // 64 k-tiles
  for (int t = 0; t < KT; ++t) {
    const int buf = t & 1;

    if (t + 1 < KT) { issue(t + 1, buf ^ 1); wait_async_20(); }
    else            { wait_async_0(); }
    __syncthreads();   // all waves' share of tile t landed in LDS

    // ---- build bf16 fragments (ISA VGPR layouts) from f32 LDS ----
    union FAB { v16bf v; __bf16 e[16]; };
    FAB a[2], bb[4];
#pragma unroll
    for (int tm = 0; tm < 2; ++tm) {
      const float* arow = &Af[buf][waveM * 32 + tm * 16 + lo][0];
      float q[16];
      // A 16-bit 16x32: lane half 'hi' -> K spans {hi*8..hi*8+7} and {16+hi*8..+7}
      *(float4*)&q[0]  = *(const float4*)(arow + hi * 8);
      *(float4*)&q[4]  = *(const float4*)(arow + hi * 8 + 4);
      *(float4*)&q[8]  = *(const float4*)(arow + 16 + hi * 8);
      *(float4*)&q[12] = *(const float4*)(arow + 16 + hi * 8 + 4);
#pragma unroll
      for (int j = 0; j < 16; ++j) a[tm].e[j] = (__bf16)q[j];
    }
#pragma unroll
    for (int tn = 0; tn < 4; ++tn) {
      const float* brow = &Bf[buf][waveN * 64 + tn * 16 + lo][0];
      float q[16];
      // B 16-bit 32x16: lane half 'hi' -> K spans {hi*8..hi*8+7} and {16+hi*8..+7} (matches A)
      *(float4*)&q[0]  = *(const float4*)(brow + hi * 8);
      *(float4*)&q[4]  = *(const float4*)(brow + hi * 8 + 4);
      *(float4*)&q[8]  = *(const float4*)(brow + 16 + hi * 8);
      *(float4*)&q[12] = *(const float4*)(brow + 16 + hi * 8 + 4);
#pragma unroll
      for (int j = 0; j < 16; ++j) bb[tn].e[j] = (__bf16)q[j];
    }

    // ---- 8 WMMAs: 16x16x32 bf16 -> f32 ----
#pragma unroll
    for (int tm = 0; tm < 2; ++tm)
#pragma unroll
      for (int tn = 0; tn < 4; ++tn)
        acc[tm][tn] = __builtin_amdgcn_wmma_f32_16x16x32_bf16(
            false, a[tm].v, false, bb[tn].v, (short)0, acc[tm][tn], false, false);

    __syncthreads();   // reads of 'buf' done before next iter DMAs into it
  }

  // ---- fused epilogue: lin + bias_lin - EPS*y*beta_norm + one ----
#pragma unroll
  for (int tm = 0; tm < 2; ++tm) {
#pragma unroll
    for (int tn = 0; tn < 4; ++tn) {
      const int col = colBase + waveN * 64 + tn * 16 + lo;
      const float bn = prec[col];
      const float ov = prec[N_OUT + col];
      const float bl = bias_lin[col];
      const int row0 = rowBase + waveM * 32 + tm * 16 + hi * 8;  // C/D: M = r + 8*hi
      union { v8f v; float f[8]; } c; c.v = acc[tm][tn];
#pragma unroll
      for (int r = 0; r < 8; ++r) {
        const size_t idx = (size_t)(row0 + r) * N_OUT + col;
        out[idx] = c.f[r] + bl - EPSILON * y[idx] * bn + ov;
      }
    }
  }
}

// ---------------- host entry ----------------
extern "C" void kernel_launch(void* const* d_in, const int* in_sizes, int n_in,
                              void* d_out, int out_size, void* d_ws, size_t ws_size,
                              hipStream_t stream) {
  const float* x        = (const float*)d_in[0];
  const float* y        = (const float*)d_in[1];
  const float* beta     = (const float*)d_in[2];
  const float* bias_lin = (const float*)d_in[3];
  // d_in[4] = W1 (dead in adv path), d_in[6] = bias1 (dead)
  const float* W2       = (const float*)d_in[5];
  const float* bias2    = (const float*)d_in[7];
  float* ws  = (float*)d_ws;   // [0..1023]=beta_norm, [1024..2047]=one
  float* out = (float*)d_out;

  beta_norm_kernel<<<(N_OUT + 255) / 256, 256, 0, stream>>>(beta, ws);
  one_kernel<<<N_OUT, 256, 0, stream>>>(W2, bias2, ws);

  dim3 grid(N_OUT / BN, BATCH / BM);   // 8 x 32 = 256 workgroups
  adv_gemm_kernel<<<grid, 256, 0, stream>>>(x, y, beta, bias_lin, ws, out);
}
